// TABGNNFusedHead_73770358276766
// MI455X (gfx1250) — matrix operand order repaired and checked by reading, not compiled, not run on hardware
//
#include <hip/hip_runtime.h>
#include <cstdint>
#include <cstddef>

// ============================================================================
// TabGNN fused head for MI455X (gfx1250, wave32, WMMA).
//
// Compute plan (see analysis): ~250 GFLOP of GEMM -> v_wmma_f32_16x16x32_f16
// with f32 accumulation. Weights (<10MB as f16) stay resident in the 192MB L2,
// so the GEMM kernel uses direct global loads shaped exactly like the CDNA5
// WMMA fragment layouts (A: lane=row, two 16B K-chunks; B: lane=column, 16 strided
// K-rows). Non-GEMM work (9-token attention, layernorm, PNA segment stats) is
// wave32 VALU with LDS staging / __shfl_xor reductions / f32+monotone-uint
// atomics.
//
// Input pointer decoding assumes JAX pytree flattening (sorted dict keys) of
// the setup_inputs() dict -- 106 leaves, enumerated below.
// ============================================================================

#define CC       128
#define NCOLS    8
#define NEDGE    100000
#define NNODE    10000
#define NTGT     4096
#define EC       4000            // edge-seq chunk: multiple of 16, divides NEDGE
#define NCHUNK   (NEDGE / EC)
#define ROWS_MAX 36864           // max token rows (4096*9) in transformer scratch

typedef __attribute__((ext_vector_type(16))) _Float16 v16h;
typedef __attribute__((ext_vector_type(8)))  float    v8f;

union VH { v16h v; uint4 q[2]; };

// ---------------------------------------------------------------------------
// Generic WMMA GEMM:  O[M,N] = act(A[M,K](f16) * B[K,N](f16) + bias[N])
// One wave per block; wave computes a 16x64 output stripe (4 WMMA tiles).
// Requires M%16==0, K%32==0, N%64==0 (true for every GEMM in this model).
// ---------------------------------------------------------------------------
__global__ void __launch_bounds__(32)
k_gemm(const _Float16* __restrict__ A, const _Float16* __restrict__ B,
       const float* __restrict__ bias, _Float16* __restrict__ Oh,
       float* __restrict__ Of, int M, int N, int K, int act)
{
    const int lane = threadIdx.x & 31;
    const int tm   = blockIdx.x;          // 16-row tile index
    const int n0   = blockIdx.y * 64;     // 64-col block
    const int r    = lane & 15;
    const int hlf  = lane >> 4;

    // A fragment: lane holds row (tm*16+r); K-chunks at hlf*8 and 16+hlf*8.
    const _Float16* arow = A + (size_t)(tm * 16 + r) * K + hlf * 8;
    // B fragment: lane holds column (n0+r); K-rows hlf*8+s (s<8) and 16+hlf*8+(s-8).
    const _Float16* brow = B + (size_t)(hlf * 8) * N + n0 + r;

    v8f acc0 = {}, acc1 = {}, acc2 = {}, acc3 = {};

    for (int k = 0; k < K; k += 32) {
        VH a;
        a.q[0] = *(const uint4*)(arow + k);
        a.q[1] = *(const uint4*)(arow + k + 16);
        const _Float16* bk = brow + (size_t)k * N;
        VH b0, b1, b2, b3;
#pragma unroll
        for (int s = 0; s < 16; ++s) {
            size_t o = (size_t)(s < 8 ? s : s + 8) * N;
            b0.v[s] = bk[o]; b1.v[s] = bk[o + 16]; b2.v[s] = bk[o + 32]; b3.v[s] = bk[o + 48];
        }
        __builtin_prefetch(arow + k + 64, 0, 0);   // -> global_prefetch_b8
        acc0 = __builtin_amdgcn_wmma_f32_16x16x32_f16(false, a.v, false, b0.v, (short)0, acc0, false, false);
        acc1 = __builtin_amdgcn_wmma_f32_16x16x32_f16(false, a.v, false, b1.v, (short)0, acc1, false, false);
        acc2 = __builtin_amdgcn_wmma_f32_16x16x32_f16(false, a.v, false, b2.v, (short)0, acc2, false, false);
        acc3 = __builtin_amdgcn_wmma_f32_16x16x32_f16(false, a.v, false, b3.v, (short)0, acc3, false, false);
    }

    // D layout: acc[i] holds row (hlf*8 + i), column n = r within the tile.
    auto epi = [&](v8f av, int t) {
        int n = n0 + t * 16 + r;
        float bi = bias ? bias[n] : 0.f;
#pragma unroll
        for (int i = 0; i < 8; ++i) {
            float v = av[i] + bi;
            if (act == 1)      v = v > 0.f ? v : 0.f;          // relu
            else if (act == 2) v = v > 0.f ? v : 0.01f * v;    // leaky relu
            size_t off = (size_t)(tm * 16 + hlf * 8 + i) * N + n;
            if (Oh) Oh[off] = (_Float16)v;
            if (Of) Of[off] = v;
        }
    };
    epi(acc0, 0); epi(acc1, 1); epi(acc2, 2); epi(acc3, 3);
}

// ---------------------------------------------------------------------------
// f32 -> f16 weight conversion (layout preserved: B is consumed as (K,N)).
// ---------------------------------------------------------------------------
__global__ void k_f32h(const float* __restrict__ in, _Float16* __restrict__ out, long n)
{
    long i = (long)blockIdx.x * blockDim.x + threadIdx.x;
    if (i < n) out[i] = (_Float16)in[i];
}

// ---------------------------------------------------------------------------
// Token encoder: token0 = cls, token(1+j) = attr[b,j]*enc_w[j,:] + enc_b[j,:]
// ---------------------------------------------------------------------------
__global__ void k_encode(const float* __restrict__ attr, const float* __restrict__ cls,
                         const float* __restrict__ enc_w, const float* __restrict__ enc_b,
                         _Float16* __restrict__ tok, int nseq, int seq0)
{
    long idx = (long)blockIdx.x * blockDim.x + threadIdx.x;
    if (idx >= (long)nseq * 9 * CC) return;
    int b = (int)(idx / (9 * CC));
    int rem = (int)(idx % (9 * CC));
    int t = rem / CC, c = rem % CC;
    float v;
    if (t == 0) v = cls[c];
    else {
        int j = t - 1;
        v = attr[(size_t)(seq0 + b) * NCOLS + j] * enc_w[j * CC + c] + enc_b[j * CC + c];
    }
    tok[idx] = (_Float16)v;
}

// ---------------------------------------------------------------------------
// Multi-head attention over 9 tokens, 8 heads, HD=16. One wave per sequence,
// K/V staged in LDS (8 waves/block -> 36KB of the 320KB/WGP).
// qkv row layout per token: [q(128) | k(128) | v(128)].
// ---------------------------------------------------------------------------
__global__ void __launch_bounds__(256)
k_attn(const _Float16* __restrict__ qkv, _Float16* __restrict__ out, int nseq)
{
    __shared__ _Float16 kv[8][2][9][CC];
    const int w = threadIdx.x >> 5, lane = threadIdx.x & 31;
    const int seq = blockIdx.x * 8 + w;
    const bool active = seq < nseq;
    const _Float16* base = qkv + (size_t)seq * 9 * 384;

    if (active) {
        for (int t = 0; t < 9; ++t) {
            *(uint2*)&kv[w][0][t][lane * 4] = *(const uint2*)(base + t * 384 + 128 + lane * 4);
            *(uint2*)&kv[w][1][t][lane * 4] = *(const uint2*)(base + t * 384 + 256 + lane * 4);
        }
    }
    __syncthreads();
    if (!active) return;

    for (int p = 0; p < 3; ++p) {
        int idx = p * 32 + lane;            // (head, q-token) pairs: 8*9 = 72
        if (idx >= 72) continue;
        int h = idx / 9, q = idx % 9;
        const _Float16* qp = base + q * 384 + h * 16;
        float qv[16];
#pragma unroll
        for (int d = 0; d < 16; ++d) qv[d] = (float)qp[d];
        float s[9], mx = -1e30f;
#pragma unroll
        for (int t = 0; t < 9; ++t) {
            float a = 0.f;
#pragma unroll
            for (int d = 0; d < 16; ++d) a += qv[d] * (float)kv[w][0][t][h * 16 + d];
            s[t] = a * 0.25f;               // 1/sqrt(HD)
            mx = fmaxf(mx, s[t]);
        }
        float den = 0.f;
#pragma unroll
        for (int t = 0; t < 9; ++t) { s[t] = __expf(s[t] - mx); den += s[t]; }
        float inv = 1.f / den;
        _Float16* op = out + ((size_t)seq * 9 + q) * CC + h * 16;
#pragma unroll
        for (int d = 0; d < 16; ++d) {
            float a = 0.f;
#pragma unroll
            for (int t = 0; t < 9; ++t) a += s[t] * (float)kv[w][1][t][h * 16 + d];
            op[d] = (_Float16)(a * inv);
        }
    }
}

// ---------------------------------------------------------------------------
// Row layernorm: O = LN(X (+R)) * g + b.  One wave32 per row, shfl reduction.
// ---------------------------------------------------------------------------
template <int D>
__global__ void k_rowln(const float* __restrict__ X, const _Float16* __restrict__ R,
                        const float* __restrict__ g, const float* __restrict__ b,
                        _Float16* __restrict__ Oh, float* __restrict__ Of,
                        int rows, long sX, long sR, long sO)
{
    constexpr int E = D / 32;
    int row = blockIdx.x * 8 + (threadIdx.x >> 5);
    if (row >= rows) return;
    int lane = threadIdx.x & 31;
    const float* x = X + (size_t)row * sX;
    float v[E], s = 0.f, s2 = 0.f;
#pragma unroll
    for (int i = 0; i < E; ++i) {
        int j = lane + i * 32;
        float f = x[j];
        if (R) f += (float)R[(size_t)row * sR + j];
        v[i] = f; s += f; s2 += f * f;
    }
#pragma unroll
    for (int m = 16; m >= 1; m >>= 1) { s += __shfl_xor(s, m, 32); s2 += __shfl_xor(s2, m, 32); }
    float mean = s / D;
    float inv = rsqrtf(s2 / D - mean * mean + 1e-5f);
#pragma unroll
    for (int i = 0; i < E; ++i) {
        int j = lane + i * 32;
        float o = (v[i] - mean) * inv * g[j] + b[j];
        if (Oh) Oh[(size_t)row * sO + j] = (_Float16)o;
        if (Of) Of[(size_t)row * sO + j] = o;
    }
}

// out = (xf + LN(h)*g + b) / 2   (final fusion head, D=384, f32 output)
__global__ void k_fuse_final(const float* __restrict__ h, const float* __restrict__ xf,
                             const float* __restrict__ g, const float* __restrict__ b,
                             float* __restrict__ out, int rows)
{
    int row = blockIdx.x * 8 + (threadIdx.x >> 5);
    if (row >= rows) return;
    int lane = threadIdx.x & 31;
    const float* x = h + (size_t)row * 384;
    float v[12], s = 0.f, s2 = 0.f;
#pragma unroll
    for (int i = 0; i < 12; ++i) { float f = x[lane + i * 32]; v[i] = f; s += f; s2 += f * f; }
#pragma unroll
    for (int m = 16; m >= 1; m >>= 1) { s += __shfl_xor(s, m, 32); s2 += __shfl_xor(s2, m, 32); }
    float mean = s / 384.f;
    float inv = rsqrtf(s2 / 384.f - mean * mean + 1e-5f);
#pragma unroll
    for (int i = 0; i < 12; ++i) {
        int j = lane + i * 32;
        float ln = (v[i] - mean) * inv * g[j] + b[j];
        out[(size_t)row * 384 + j] = (xf[(size_t)row * 384 + j] + ln) * 0.5f;
    }
}

// ---------------------------------------------------------------------------
// GNN support kernels
// ---------------------------------------------------------------------------
__global__ void k_node(const float* __restrict__ x, const float* __restrict__ nw,
                       const float* __restrict__ nb, float* __restrict__ xg,
                       _Float16* __restrict__ xgh)
{
    int idx = blockIdx.x * blockDim.x + threadIdx.x;
    if (idx >= NNODE * CC) return;
    int n = idx >> 7, c = idx & 127;
    float v = x[n] * nw[c] + nb[c];
    xg[idx] = v; xgh[idx] = (_Float16)v;
}

__global__ void k_count(const int* __restrict__ dst, float* __restrict__ cnt)
{
    int i = blockIdx.x * blockDim.x + threadIdx.x;
    if (i < NEDGE) atomicAdd(&cnt[dst[i]], 1.f);
}

__global__ void k_logdeg(const float* __restrict__ cnt, float* __restrict__ logsum)
{
    int i = blockIdx.x * blockDim.x + threadIdx.x;
    if (i < NNODE) atomicAdd(logsum, logf(cnt[i] + 1.f));
}

// concat [x[dst], x[src], ee] -> A_pre (E x 384 f16)
__global__ void k_pre(const _Float16* __restrict__ xgh, const int* __restrict__ src,
                      const int* __restrict__ dst, const _Float16* __restrict__ ee,
                      _Float16* __restrict__ ap)
{
    long idx = (long)blockIdx.x * blockDim.x + threadIdx.x;
    if (idx >= (long)NEDGE * 384) return;
    int e = (int)(idx / 384), c = (int)(idx % 384);
    _Float16 v;
    if (c < CC)            v = xgh[(size_t)dst[e] * CC + c];
    else if (c < 2 * CC)   v = xgh[(size_t)src[e] * CC + (c - CC)];
    else                   v = ee[(size_t)e * CC + (c - 2 * CC)];
    ap[idx] = v;
}

// order-preserving float <-> uint for atomic max/min
__device__ inline unsigned f2mono(float f) {
    unsigned u = __float_as_uint(f);
    return (u & 0x80000000u) ? ~u : (u | 0x80000000u);
}
__device__ inline float mono2f(unsigned u) {
    return __uint_as_float((u & 0x80000000u) ? (u & 0x7FFFFFFFu) : ~u);
}

__global__ void k_stats(const float* __restrict__ m, const int* __restrict__ dst,
                        float* __restrict__ ssum, float* __restrict__ ssq,
                        unsigned* __restrict__ smx, unsigned* __restrict__ smn)
{
    long idx = (long)blockIdx.x * blockDim.x + threadIdx.x;
    if (idx >= (long)NEDGE * CC) return;
    int e = (int)(idx >> 7), c = (int)(idx & 127);
    float v = m[idx];
    size_t o = (size_t)dst[e] * CC + c;
    atomicAdd(&ssum[o], v);
    atomicAdd(&ssq[o], v * v);
    atomicMax(&smx[o], f2mono(v));
    atomicMin(&smn[o], f2mono(v));
}

// build A_post (N x 1664 f16) = [x, agg, agg*amp, agg/amp], agg=[mean,mx,mn,std]
__global__ void k_post(const float* __restrict__ xg, const float* __restrict__ cnt,
                       const float* __restrict__ logsum, const float* __restrict__ ssum,
                       const float* __restrict__ ssq, const unsigned* __restrict__ smx,
                       const unsigned* __restrict__ smn, _Float16* __restrict__ ap)
{
    int idx = blockIdx.x * blockDim.x + threadIdx.x;
    if (idx >= NNODE * CC) return;
    int n = idx >> 7, c = idx & 127;
    float cn = cnt[n];
    float c1 = fmaxf(cn, 1.f);
    float mean = ssum[idx] / c1;
    float mean2 = ssq[idx] / c1;
    float sd = sqrtf(fmaxf(mean2 - mean * mean, 0.f) + 1e-5f);
    float mx = cn > 0.f ? mono2f(smx[idx]) : 0.f;
    float mn = cn > 0.f ? mono2f(smn[idx]) : 0.f;
    float lda = logsum[0] * (1.f / NNODE);
    float amp = logf(c1 + 1.f) / lda;
    float ia = 1.f / amp;
    _Float16* row = ap + (size_t)n * 1664;
    row[c]          = (_Float16)xg[idx];
    row[128 + c]    = (_Float16)mean;        row[256 + c]  = (_Float16)mx;
    row[384 + c]    = (_Float16)mn;          row[512 + c]  = (_Float16)sd;
    row[640 + c]    = (_Float16)(mean * amp); row[768 + c]  = (_Float16)(mx * amp);
    row[896 + c]    = (_Float16)(mn * amp);  row[1024 + c] = (_Float16)(sd * amp);
    row[1152 + c]   = (_Float16)(mean * ia); row[1280 + c] = (_Float16)(mx * ia);
    row[1408 + c]   = (_Float16)(mn * ia);   row[1536 + c] = (_Float16)(sd * ia);
}

__global__ void k_bnstat(const float* __restrict__ h, float* __restrict__ bs,
                         float* __restrict__ bs2)
{
    int idx = blockIdx.x * blockDim.x + threadIdx.x;
    if (idx >= NNODE * CC) return;
    int c = idx & 127;
    float v = h[idx];
    atomicAdd(&bs[c], v);
    atomicAdd(&bs2[c], v * v);
}

__global__ void k_bnapply(const float* __restrict__ h, const float* __restrict__ bs,
                          const float* __restrict__ bs2, const float* __restrict__ g,
                          const float* __restrict__ b, float* __restrict__ xg,
                          _Float16* __restrict__ xgh)
{
    int idx = blockIdx.x * blockDim.x + threadIdx.x;
    if (idx >= NNODE * CC) return;
    int c = idx & 127;
    float mu = bs[c] * (1.f / NNODE);
    float var = bs2[c] * (1.f / NNODE) - mu * mu;
    float y = (h[idx] - mu) * rsqrtf(var + 1e-5f) * g[c] + b[c];
    y = fmaxf(y, 0.f);
    float o = (xg[idx] + y) * 0.5f;
    xg[idx] = o; xgh[idx] = (_Float16)o;
}

// xf = [x_tab[:,0,:], x_gnn[tei0], x_gnn[tei1]]  (4096 x 384 f32)
__global__ void k_build_xf(const _Float16* __restrict__ xtab, const float* __restrict__ xg,
                           const int* __restrict__ tei, float* __restrict__ xf)
{
    long idx = (long)blockIdx.x * blockDim.x + threadIdx.x;
    if (idx >= (long)NTGT * 384) return;
    int i = (int)(idx / 384), j = (int)(idx % 384);
    float v;
    if (j < CC)            v = (float)xtab[(size_t)i * 9 * CC + j];
    else if (j < 2 * CC)   v = xg[(size_t)tei[i] * CC + (j - CC)];
    else                   v = xg[(size_t)tei[NTGT + i] * CC + (j - 2 * CC)];
    xf[idx] = v;
}

// ============================================================================
// Host orchestration
// ============================================================================
struct Bump {
    char* base; size_t off;
    void* take(size_t n) { void* p = base + off; off += (n + 255) & ~(size_t)255; return p; }
};

static inline void gemm(hipStream_t s, const _Float16* A, const _Float16* B,
                        const float* bias, _Float16* Oh, float* Of,
                        int M, int N, int K, int act)
{
    dim3 g(M / 16, N / 64);
    k_gemm<<<g, 32, 0, s>>>(A, B, bias, Oh, Of, M, N, K, act);
}

static const _Float16* cvt(const void* w, size_t n, Bump& ws, hipStream_t s)
{
    _Float16* o = (_Float16*)ws.take(n * 2);
    k_f32h<<<(int)((n + 255) / 256), 256, 0, s>>>((const float*)w, o, (long)n);
    return o;
}

// ---- input leaf indices: JAX pytree (sorted-key) flatten of setup_inputs() ----
enum {
    IN_EDGE_ATTR = 0, IN_EDGE_INDEX = 1,
    P_CLS = 2, P_ENC_B = 3, P_ENC_W = 4,
    FZ = 5,                       // fuse block (34 leaves: 5..38)
    HT = 39,                      // head_tab block (12 leaves: 39..50)
    P_HEAD_TABNORM_B = 51, P_HEAD_TABNORM_G = 52,
    LY0 = 53, LY1 = 77,           // layer blocks (24 leaves each)
    P_NODE_B = 101, P_NODE_W = 102,
    IN_TGT_ATTR = 103, IN_TGT_INDEX = 104, IN_X = 105
};
// member offsets within blocks (sorted keys)
enum { TO_B1 = 0, TO_B2, TO_BOUT, TO_BQKV, TO_LN1B, TO_LN1G, TO_LN2B, TO_LN2G,
       TO_W1, TO_W2, TO_WOUT, TO_WQKV };
enum { LO_BN_B = 0, LO_BN_G, LO_EDGE_B, LO_EDGE_W, LO_LIN_B, LO_LIN_W,
       LO_POST_B, LO_POST_W, LO_PRE_B, LO_PRE_W, LO_TAB,
       LO_TABNORM_B = LO_TAB + 12, LO_TABNORM_G };
enum { FO_BN_B = 0, FO_BN_G, FO_EDGE_B, FO_EDGE_W, FO_FB1, FO_FB2, FO_FB3,
       FO_FW1, FO_FW2, FO_FW3, FO_FUSE_LN_B, FO_FUSE_LN_G, FO_FUSENORM_B,
       FO_FUSENORM_G, FO_LIN_B, FO_LIN_W, FO_POST_B, FO_POST_W, FO_PRE_B,
       FO_PRE_W, FO_TAB, FO_TABNORM_B = FO_TAB + 12, FO_TABNORM_G };

struct TabW {
    const _Float16 *wqkv, *wout, *w1, *w2;
    const float *bqkv, *bout, *b1, *b2, *ln1g, *ln1b, *ln2g, *ln2b;
};
struct LayerW {
    TabW tab;
    const float *tabnorm_g, *tabnorm_b;
    const _Float16 *edge_w, *pre_w, *post_w, *lin_w;
    const float *edge_b, *pre_b, *post_b, *lin_b, *bn_g, *bn_b;
};

static TabW make_tab(void* const* din, int base, Bump& ws, hipStream_t s)
{
    TabW w;
    w.b1   = (const float*)din[base + TO_B1];
    w.b2   = (const float*)din[base + TO_B2];
    w.bout = (const float*)din[base + TO_BOUT];
    w.bqkv = (const float*)din[base + TO_BQKV];
    w.ln1b = (const float*)din[base + TO_LN1B];
    w.ln1g = (const float*)din[base + TO_LN1G];
    w.ln2b = (const float*)din[base + TO_LN2B];
    w.ln2g = (const float*)din[base + TO_LN2G];
    w.w1   = cvt(din[base + TO_W1],   CC * CC,      ws, s);
    w.w2   = cvt(din[base + TO_W2],   CC * CC,      ws, s);
    w.wout = cvt(din[base + TO_WOUT], CC * CC,      ws, s);
    w.wqkv = cvt(din[base + TO_WQKV], CC * 3 * CC,  ws, s);
    return w;
}

static LayerW make_layer(void* const* din, int b, Bump& ws, hipStream_t s)
{
    LayerW L;
    L.bn_b   = (const float*)din[b + LO_BN_B];
    L.bn_g   = (const float*)din[b + LO_BN_G];
    L.edge_b = (const float*)din[b + LO_EDGE_B];
    L.edge_w = cvt(din[b + LO_EDGE_W], CC * CC, ws, s);
    L.lin_b  = (const float*)din[b + LO_LIN_B];
    L.lin_w  = cvt(din[b + LO_LIN_W], CC * CC, ws, s);
    L.post_b = (const float*)din[b + LO_POST_B];
    L.post_w = cvt(din[b + LO_POST_W], 1664 * CC, ws, s);
    L.pre_b  = (const float*)din[b + LO_PRE_B];
    L.pre_w  = cvt(din[b + LO_PRE_W], 384 * CC, ws, s);
    L.tab    = make_tab(din, b + LO_TAB, ws, s);
    L.tabnorm_b = (const float*)din[b + LO_TABNORM_B];
    L.tabnorm_g = (const float*)din[b + LO_TABNORM_G];
    return L;
}

static LayerW make_fuse(void* const* din, Bump& ws, hipStream_t s)
{
    LayerW L;
    L.bn_b   = (const float*)din[FZ + FO_BN_B];
    L.bn_g   = (const float*)din[FZ + FO_BN_G];
    L.edge_b = (const float*)din[FZ + FO_EDGE_B];
    L.edge_w = cvt(din[FZ + FO_EDGE_W], CC * CC, ws, s);
    L.lin_b  = (const float*)din[FZ + FO_LIN_B];
    L.lin_w  = cvt(din[FZ + FO_LIN_W], CC * CC, ws, s);
    L.post_b = (const float*)din[FZ + FO_POST_B];
    L.post_w = cvt(din[FZ + FO_POST_W], 1664 * CC, ws, s);
    L.pre_b  = (const float*)din[FZ + FO_PRE_B];
    L.pre_w  = cvt(din[FZ + FO_PRE_W], 384 * CC, ws, s);
    L.tab    = make_tab(din, FZ + FO_TAB, ws, s);
    L.tabnorm_b = (const float*)din[FZ + FO_TABNORM_B];
    L.tabnorm_g = (const float*)din[FZ + FO_TABNORM_G];
    return L;
}

// One transformer layer on nseq 9-token sequences. Leaves post-ln2 f32 tokens
// in t_f32; `tok` buffer is clobbered.
static void run_tab(hipStream_t s, const TabW& W, _Float16* tok, int nseq,
                    _Float16* t_qkv, _Float16* t_att, float* t_f32)
{
    int M = nseq * 9;
    gemm(s, tok, W.wqkv, W.bqkv, t_qkv, nullptr, M, 384, CC, 0);
    k_attn<<<nseq / 8, 256, 0, s>>>(t_qkv, t_att, nseq);
    gemm(s, t_att, W.wout, W.bout, nullptr, t_f32, M, CC, CC, 0);
    k_rowln<128><<<(M + 7) / 8, 256, 0, s>>>(t_f32, tok, W.ln1g, W.ln1b,
                                             t_att, nullptr, M, CC, CC, CC);
    gemm(s, t_att, W.w1, W.b1, tok, nullptr, M, CC, CC, 1);
    gemm(s, tok, W.w2, W.b2, nullptr, t_f32, M, CC, CC, 0);
    k_rowln<128><<<(M + 7) / 8, 256, 0, s>>>(t_f32, t_att, W.ln2g, W.ln2b,
                                             nullptr, t_f32, M, CC, CC, CC);
}

extern "C" void kernel_launch(void* const* d_in, const int* in_sizes, int n_in,
                              void* d_out, int out_size, void* d_ws, size_t ws_size,
                              hipStream_t stream)
{
    (void)in_sizes; (void)n_in; (void)out_size;
    // Total workspace need ~= 375 MB (weights f16 + activations). Bail safely
    // (memory-safe no-op) if the harness scratch is smaller.
    if (ws_size < (size_t)400 * 1024 * 1024) return;

    Bump ws{(char*)d_ws, 0};

    // ---- raw inputs ----
    const float* edge_attr = (const float*)d_in[IN_EDGE_ATTR];
    const int*   eidx      = (const int*)d_in[IN_EDGE_INDEX];
    const int*   src       = eidx;
    const int*   dst       = eidx + NEDGE;
    const float* cls       = (const float*)d_in[P_CLS];
    const float* enc_b     = (const float*)d_in[P_ENC_B];
    const float* enc_w     = (const float*)d_in[P_ENC_W];
    const float* tgt_attr  = (const float*)d_in[IN_TGT_ATTR];
    const int*   tei       = (const int*)d_in[IN_TGT_INDEX];
    const float* xin       = (const float*)d_in[IN_X];
    const float* node_b    = (const float*)d_in[P_NODE_B];
    const float* node_w    = (const float*)d_in[P_NODE_W];
    const float* ht_norm_b = (const float*)d_in[P_HEAD_TABNORM_B];
    const float* ht_norm_g = (const float*)d_in[P_HEAD_TABNORM_G];
    const float* fuse_ln_b   = (const float*)d_in[FZ + FO_FUSE_LN_B];
    const float* fuse_ln_g   = (const float*)d_in[FZ + FO_FUSE_LN_G];
    const float* fusenorm_b  = (const float*)d_in[FZ + FO_FUSENORM_B];
    const float* fusenorm_g  = (const float*)d_in[FZ + FO_FUSENORM_G];
    const float* fb1 = (const float*)d_in[FZ + FO_FB1];
    const float* fb2 = (const float*)d_in[FZ + FO_FB2];
    const float* fb3 = (const float*)d_in[FZ + FO_FB3];

    // ---- weight conversion (f32 -> f16, layout kept (K,N)) ----
    TabW headW = make_tab(d_in, HT, ws, stream);
    LayerW L[3];
    L[0] = make_layer(d_in, LY0, ws, stream);
    L[1] = make_layer(d_in, LY1, ws, stream);
    L[2] = make_fuse(d_in, ws, stream);
    const _Float16* fw1 = cvt(d_in[FZ + FO_FW1], 384 * 1536, ws, stream);
    const _Float16* fw2 = cvt(d_in[FZ + FO_FW2], (size_t)1536 * 1536, ws, stream);
    const _Float16* fw3 = cvt(d_in[FZ + FO_FW3], 1536 * 384, ws, stream);

    // ---- activation buffers ----
    float*    x_gnn   = (float*)   ws.take((size_t)NNODE * CC * 4);
    _Float16* x_gnn_h = (_Float16*)ws.take((size_t)NNODE * CC * 2);
    _Float16* e16     = (_Float16*)ws.take((size_t)NEDGE * CC * 2);
    _Float16* x_tab   = (_Float16*)ws.take((size_t)NTGT * 9 * CC * 2);
    float*    cnt     = (float*)   ws.take((size_t)NNODE * 4);
    float*    logsum  = (float*)   ws.take(256);
    _Float16* t_x     = (_Float16*)ws.take((size_t)ROWS_MAX * CC * 2);
    _Float16* t_qkv   = (_Float16*)ws.take((size_t)ROWS_MAX * 384 * 2);
    _Float16* t_att   = (_Float16*)ws.take((size_t)ROWS_MAX * CC * 2);
    float*    t_f32   = (float*)   ws.take((size_t)ROWS_MAX * CC * 4);
    _Float16* ee16    = (_Float16*)ws.take((size_t)NEDGE * CC * 2);
    _Float16* a_pre   = (_Float16*)ws.take((size_t)NEDGE * 384 * 2);
    float*    m_f32   = (float*)   ws.take((size_t)NEDGE * CC * 4);
    float*    ssum    = (float*)   ws.take((size_t)NNODE * CC * 4);
    float*    ssq     = (float*)   ws.take((size_t)NNODE * CC * 4);
    unsigned* smx     = (unsigned*)ws.take((size_t)NNODE * CC * 4);
    unsigned* smn     = (unsigned*)ws.take((size_t)NNODE * CC * 4);
    _Float16* a_post  = (_Float16*)ws.take((size_t)NNODE * 1664 * 2);
    _Float16* h1      = (_Float16*)ws.take((size_t)NNODE * CC * 2);
    float*    hbuf    = (float*)   ws.take((size_t)NNODE * CC * 4);
    float*    bns     = (float*)   ws.take(512);
    float*    bns2    = (float*)   ws.take(512);
    float*    xf      = (float*)   ws.take((size_t)NTGT * 384 * 4);
    _Float16* a1      = (_Float16*)ws.take((size_t)NTGT * 384 * 2);
    _Float16* a2      = (_Float16*)ws.take((size_t)NTGT * 1536 * 2);
    _Float16* a3      = (_Float16*)ws.take((size_t)NTGT * 1536 * 2);
    float*    hf      = (float*)   ws.take((size_t)NTGT * 384 * 4);

    const size_t STAT_BYTES = (size_t)NNODE * CC * 4;

    // ---- node embedding + degree stats ----
    k_node<<<(NNODE * CC + 255) / 256, 256, 0, stream>>>(xin, node_w, node_b, x_gnn, x_gnn_h);
    hipMemsetAsync(cnt, 0, (size_t)NNODE * 4, stream);
    hipMemsetAsync(logsum, 0, 4, stream);
    k_count<<<(NEDGE + 255) / 256, 256, 0, stream>>>(dst, cnt);
    k_logdeg<<<(NNODE + 255) / 256, 256, 0, stream>>>(cnt, logsum);

    // ---- x_tab token encoding (4096 target sequences) ----
    k_encode<<<((long)NTGT * 9 * CC + 255) / 256, 256, 0, stream>>>(
        tgt_attr, cls, enc_w, enc_b, x_tab, NTGT, 0);

    // ---- head transformer over all edges (chunked) -> e (E x 128 f16) ----
    for (int c = 0; c < NCHUNK; ++c) {
        k_encode<<<((long)EC * 9 * CC + 255) / 256, 256, 0, stream>>>(
            edge_attr, cls, enc_w, enc_b, t_x, EC, c * EC);
        run_tab(stream, headW, t_x, EC, t_qkv, t_att, t_f32);
        // head_tabnorm on token-0 rows only (LN is row-local), strided read
        k_rowln<128><<<(EC + 7) / 8, 256, 0, stream>>>(
            t_f32, nullptr, ht_norm_g, ht_norm_b,
            e16 + (size_t)c * EC * CC, nullptr, EC, 9 * CC, 0, CC);
    }

    // ---- main layers: transformer on x_tab + PNA GNN step ----
    for (int li = 0; li < 3; ++li) {
        const LayerW& lp = L[li];

        run_tab(stream, lp.tab, x_tab, NTGT, t_qkv, t_att, t_f32);
        k_rowln<128><<<(NTGT * 9 + 7) / 8, 256, 0, stream>>>(
            t_f32, nullptr, lp.tabnorm_g, lp.tabnorm_b, x_tab, nullptr,
            NTGT * 9, CC, 0, CC);

        // gnn_step
        gemm(stream, e16, lp.edge_w, lp.edge_b, ee16, nullptr, NEDGE, CC, CC, 0);
        k_pre<<<((long)NEDGE * 384 + 255) / 256, 256, 0, stream>>>(x_gnn_h, src, dst, ee16, a_pre);
        gemm(stream, a_pre, lp.pre_w, lp.pre_b, nullptr, m_f32, NEDGE, CC, 384, 0);
        hipMemsetAsync(ssum, 0, STAT_BYTES, stream);
        hipMemsetAsync(ssq, 0, STAT_BYTES, stream);
        hipMemsetAsync(smx, 0x00, STAT_BYTES, stream);   // mono(-inf) > 0 -> identity
        hipMemsetAsync(smn, 0xFF, STAT_BYTES, stream);
        k_stats<<<((long)NEDGE * CC + 255) / 256, 256, 0, stream>>>(m_f32, dst, ssum, ssq, smx, smn);
        k_post<<<(NNODE * CC + 255) / 256, 256, 0, stream>>>(
            x_gnn, cnt, logsum, ssum, ssq, smx, smn, a_post);
        gemm(stream, a_post, lp.post_w, lp.post_b, h1, nullptr, NNODE, CC, 1664, 0);
        gemm(stream, h1, lp.lin_w, lp.lin_b, nullptr, hbuf, NNODE, CC, CC, 0);
        hipMemsetAsync(bns, 0, 512, stream);
        hipMemsetAsync(bns2, 0, 512, stream);
        k_bnstat<<<(NNODE * CC + 255) / 256, 256, 0, stream>>>(hbuf, bns, bns2);
        k_bnapply<<<(NNODE * CC + 255) / 256, 256, 0, stream>>>(
            hbuf, bns, bns2, lp.bn_g, lp.bn_b, x_gnn, x_gnn_h);
    }

    // ---- fusion head ----
    k_build_xf<<<((long)NTGT * 384 + 255) / 256, 256, 0, stream>>>(x_tab, x_gnn, tei, xf);
    k_rowln<384><<<(NTGT + 7) / 8, 256, 0, stream>>>(
        xf, nullptr, fuse_ln_g, fuse_ln_b, a1, nullptr, NTGT, 384, 0, 384);
    gemm(stream, a1, fw1, fb1, a2, nullptr, NTGT, 1536, 384, 2);
    gemm(stream, a2, fw2, fb2, a3, nullptr, NTGT, 1536, 1536, 2);
    gemm(stream, a3, fw3, fb3, nullptr, hf, NTGT, 384, 1536, 0);
    k_fuse_final<<<(NTGT + 7) / 8, 256, 0, stream>>>(
        hf, xf, fusenorm_g, fusenorm_b, (float*)d_out, NTGT);
}
